// MOELayer_12051678232646
// MI455X (gfx1250) — matrix operand order repaired and checked by reading, not compiled, not run on hardware
//
#include <hip/hip_runtime.h>
#include <stdint.h>

typedef __attribute__((ext_vector_type(16))) __bf16 v16bf;
typedef __attribute__((ext_vector_type(8)))  float  v8f;
typedef __attribute__((ext_vector_type(4)))  unsigned int u32x4;
typedef __attribute__((ext_vector_type(8)))  int i32x8;
typedef __attribute__((ext_vector_type(4)))  int i32x4;

#define N_TOK 16384
#define CDIM  1024
#define HDIM  4096
#define NEXP  8
#define CAP   5120   // int(1.25 * 16384 * 2 / 8)

// ---------------------------------------------------------------------------
// Router: logits = x @ w_g.T ; top-2 ; softmax over the two logits.
// ---------------------------------------------------------------------------
__global__ __launch_bounds__(256) void router_kernel(const float* __restrict__ x,
                                                     const float* __restrict__ wg,
                                                     int* __restrict__ top_idx,
                                                     float* __restrict__ probs) {
  __shared__ float wgs[NEXP * CDIM];
  for (int i = threadIdx.x; i < NEXP * CDIM; i += 256) wgs[i] = wg[i];
  __syncthreads();
  int wave = threadIdx.x >> 5, lane = threadIdx.x & 31;
  int n = blockIdx.x * 8 + wave;
  const float* xr = x + (size_t)n * CDIM;
  float acc[NEXP];
#pragma unroll
  for (int e = 0; e < NEXP; e++) acc[e] = 0.f;
  for (int c = lane; c < CDIM; c += 32) {
    float xv = xr[c];
#pragma unroll
    for (int e = 0; e < NEXP; e++) acc[e] += xv * wgs[e * CDIM + c];
  }
#pragma unroll
  for (int off = 16; off >= 1; off >>= 1) {
#pragma unroll
    for (int e = 0; e < NEXP; e++) acc[e] += __shfl_down(acc[e], off, 32);
  }
  if (lane == 0) {
    int e1 = 0; float v1 = acc[0];
#pragma unroll
    for (int e = 1; e < NEXP; e++) { if (acc[e] > v1) { v1 = acc[e]; e1 = e; } }
    int e2 = (e1 == 0) ? 1 : 0; float v2 = acc[e2];
#pragma unroll
    for (int e = 0; e < NEXP; e++) {
      if (e != e1 && acc[e] > v2) { v2 = acc[e]; e2 = e; }
    }
    float t = __expf(v2 - v1);      // v1 >= v2
    float p1 = 1.f / (1.f + t);
    float p2 = t * p1;
    top_idx[n * 2]     = e1;
    top_idx[n * 2 + 1] = e2;
    probs[n * 2]       = p1;
    probs[n * 2 + 1]   = p2;
  }
}

// ---------------------------------------------------------------------------
// Ordered rank scan over flattened (k-major, token) assignment sequence.
// ---------------------------------------------------------------------------
__global__ __launch_bounds__(1024) void rank_kernel(const int* __restrict__ top_idx,
                                                    const float* __restrict__ probs,
                                                    int* __restrict__ rank,
                                                    float* __restrict__ probs_m) {
  __shared__ int base[NEXP];
  __shared__ int wtot[32][NEXP];
  __shared__ int wbase[32][NEXP];
  int tid = threadIdx.x, lane = tid & 31, wave = tid >> 5;
  if (tid < NEXP) base[tid] = 0;
  __syncthreads();
  for (int m0 = 0; m0 < 2 * N_TOK; m0 += 1024) {
    int m = m0 + tid;
    int k = (m >= N_TOK) ? 1 : 0;
    int n = m - k * N_TOK;
    int e = top_idx[n * 2 + k];
    unsigned mymask = 0;
#pragma unroll
    for (int q = 0; q < NEXP; q++) {
      unsigned msk = (unsigned)__ballot(e == q);
      if (lane == 0) wtot[wave][q] = __popc(msk);
      if (q == e) mymask = msk;
    }
    unsigned lt = (lane == 0) ? 0u : (0xFFFFFFFFu >> (32 - lane));
    int prefix = __popc(mymask & lt);
    __syncthreads();
    if (tid < NEXP) {
      int run = base[tid];
      for (int w = 0; w < 32; w++) { wbase[w][tid] = run; run += wtot[w][tid]; }
      base[tid] = run;
    }
    __syncthreads();
    int r = wbase[wave][e] + prefix;
    rank[n * 2 + k]    = r;
    probs_m[n * 2 + k] = (r < CAP) ? probs[n * 2 + k] : 0.f;
    __syncthreads();
  }
}

// ---------------------------------------------------------------------------
__global__ __launch_bounds__(256) void zero_kernel(uint4* __restrict__ p, long long n16) {
  long long i = (long long)blockIdx.x * 256 + threadIdx.x;
  if (i < n16) p[i] = make_uint4(0u, 0u, 0u, 0u);
}

// ---------------------------------------------------------------------------
__global__ __launch_bounds__(256) void dispatch_kernel(const float* __restrict__ x,
                                                       const int* __restrict__ top_idx,
                                                       const int* __restrict__ rank,
                                                       __bf16* __restrict__ buf) {
  int m = blockIdx.x;
  int n = m >> 1, k = m & 1;
  int r = rank[n * 2 + k];
  if (r >= CAP) return;
  int e = top_idx[n * 2 + k];
  const float* src = x + (size_t)n * CDIM;
  __bf16* dst = buf + ((size_t)e * CAP + r) * CDIM;
  for (int c = threadIdx.x; c < CDIM; c += 256) dst[c] = (__bf16)src[c];
}

// ---------------------------------------------------------------------------
// Tiled transpose + f32->bf16 cast: in [E][R][Cc] -> out [E][Cc][R].
// ---------------------------------------------------------------------------
__global__ __launch_bounds__(256) void transpose_cast_kernel(const float* __restrict__ in,
                                                             __bf16* __restrict__ out,
                                                             int R, int Cc) {
  __shared__ float t[32][33];
  int e = blockIdx.z;
  const float* ine = in + (size_t)e * R * Cc;
  __bf16* oute = out + (size_t)e * R * Cc;
  int tx = threadIdx.x & 31, ty = threadIdx.x >> 5;
  int c0 = blockIdx.x * 32, r0 = blockIdx.y * 32;
#pragma unroll
  for (int i = 0; i < 4; i++)
    t[ty + i * 8][tx] = ine[(size_t)(r0 + ty + i * 8) * Cc + (c0 + tx)];
  __syncthreads();
#pragma unroll
  for (int i = 0; i < 4; i++)
    oute[(size_t)(c0 + ty + i * 8) * R + (r0 + tx)] = (__bf16)t[tx][ty + i * 8];
}

// ---------------------------------------------------------------------------
// TDM: DMA one 128-row x 32-col bf16 tile (row stride K elements) from global
// into LDS at lds_off, with native LDS padding to stride 40 bf16:
//   pad_interval=3 -> pad after every 16 DWORDs (64B = one 32-elem bf16 row)
//   pad_amount  =3 -> insert 4 DWORDs (16B = 8 bf16)       => row stride 80B
// D# per CDNA5 ISA ch.8: group0 {count, lds_addr, global_addr, type=2},
// group1 {data_size=1(2B)|pad bits, tensor_dim0=K, tensor_dim1=1M,
//         tile_dim0=32, tile_dim1=128, dim0_stride=K}, groups 2/3 zero (2D).
// This toolchain's builtin takes the 6-arg form (trailing int32x8 unused).
// ---------------------------------------------------------------------------
__device__ __forceinline__ void tdm_load_tile(const __bf16* gptr, unsigned lds_off, int K) {
  unsigned long long ga = (unsigned long long)(uintptr_t)gptr;
  unsigned uk = (unsigned)K;
  u32x4 g0 = { 1u,                                   // count=1 (valid, user mode)
               lds_off,                               // D#.lds_addr
               (unsigned)ga,                          // global_addr[31:0]
               (unsigned)(ga >> 32) | (2u << 30) };   // global_addr[56:32] | type=2
  i32x8 g1 = { (int)((1u << 16) | (1u << 20) | (3u << 22) | (3u << 25)),
               (int)(uk << 16),                       // tensor_dim0[15:0]
               (int)(uk >> 16),                       // tensor_dim0[31:16] | td1 lo16 (=0)
               (int)(16u | (32u << 16)),              // td1 hi16 (1<<20) | tile_dim0=32
               (int)128,                              // tile_dim1=128, tile_dim2=0
               (int)uk,                               // dim0_stride[31:0] = K
               0, 0 };                                // stride hi / dim1_stride unused
  i32x4 z4 = { 0, 0, 0, 0 };
  i32x8 z8 = { 0, 0, 0, 0, 0, 0, 0, 0 };
  __builtin_amdgcn_tensor_load_to_lds(g0, g1, z4, z4, z8, 0);
}

__device__ __forceinline__ unsigned lds_off_of(const void* p) {
  return (unsigned)(uintptr_t)p;    // LDS aperture: addr[31:0] is the LDS offset
}

// ---------------------------------------------------------------------------
// bf16 WMMA GEMM with TDM-staged, double-buffered LDS tiles.
// Block tile 128x128, 8 waves (2M x 4N); each wave: 4x2 grid of 16x16 tiles,
// 8x v_wmma_f32_16x16x32_bf16 per 32-deep K step.
// Wave 0 DMAs the A tile stream, wave 1 the B tile stream (TDM ignores EXEC;
// one issue per wave). 2-deep pipeline on TENSORcnt; barriers publish tiles
// and fence LDS reuse (WMMA consumption forces the dscnt wait before the
// closing barrier, so the next DMA never overwrites live fragments).
// ---------------------------------------------------------------------------
template <int ACT>
__global__ __launch_bounds__(256) void gemm_bf16_kernel(const __bf16* __restrict__ A,
                                                        const __bf16* __restrict__ Bt,
                                                        void* __restrict__ Out,
                                                        int M, int Nn, int K) {
  __shared__ __bf16 As[2][128 * 40];
  __shared__ __bf16 Bs[2][128 * 40];
  int e = blockIdx.z;
  const __bf16* Ae = A + (size_t)e * M * K;
  const __bf16* Be = Bt + (size_t)e * Nn * K;
  int m0 = blockIdx.x * 128, n0 = blockIdx.y * 128;
  int tid = threadIdx.x, lane = tid & 31, wave = tid >> 5;
  int wm = wave & 1, wn = wave >> 1;
  int ml = lane & 15, hh = lane >> 4;

  const __bf16* arow = Ae + (size_t)m0 * K;
  const __bf16* brow = Be + (size_t)n0 * K;
  int nsteps = K >> 5;

  // Prologue: issue tiles 0 and 1.
  if (wave == 0) {
    tdm_load_tile(arow + 0, lds_off_of(&As[0][0]), K);
    if (nsteps > 1) tdm_load_tile(arow + 32, lds_off_of(&As[1][0]), K);
  } else if (wave == 1) {
    tdm_load_tile(brow + 0, lds_off_of(&Bs[0][0]), K);
    if (nsteps > 1) tdm_load_tile(brow + 32, lds_off_of(&Bs[1][0]), K);
  }

  v8f acc[4][2] = {};

  for (int i = 0; i < nsteps; i++) {
    int p = i & 1;
    if (wave < 2) {                       // DMA-owning waves gate on TENSORcnt
      if (i == nsteps - 1) __builtin_amdgcn_s_wait_tensorcnt(0);
      else                 __builtin_amdgcn_s_wait_tensorcnt(1);
    }
    __syncthreads();                      // tile i visible to all waves

    union Frag { uint4 q[2]; v16bf v; };
    v16bf af[4], bfr[2];
#pragma unroll
    for (int mi = 0; mi < 4; mi++) {
      const __bf16* pr = &As[p][(wm * 64 + mi * 16 + ml) * 40];
      Frag f;
      f.q[0] = *(const uint4*)&pr[hh * 8];
      f.q[1] = *(const uint4*)&pr[16 + hh * 8];
      af[mi] = f.v;
    }
#pragma unroll
    for (int ni = 0; ni < 2; ni++) {
      const __bf16* pr = &Bs[p][(wn * 32 + ni * 16 + ml) * 40 + hh * 8];
      Frag f;
      f.q[0] = *(const uint4*)&pr[0];
      f.q[1] = *(const uint4*)&pr[16];
      bfr[ni] = f.v;
    }
#pragma unroll
    for (int mi = 0; mi < 4; mi++) {
#pragma unroll
      for (int ni = 0; ni < 2; ni++) {
        acc[mi][ni] = __builtin_amdgcn_wmma_f32_16x16x32_bf16(
            false, af[mi], false, bfr[ni], (short)0, acc[mi][ni], false, false);
      }
    }

    __syncthreads();                      // all waves done with buffer p
    if (i + 2 < nsteps) {                 // refill buffer p with tile i+2
      int k2 = (i + 2) << 5;
      if (wave == 0)      tdm_load_tile(arow + k2, lds_off_of(&As[p][0]), K);
      else if (wave == 1) tdm_load_tile(brow + k2, lds_off_of(&Bs[p][0]), K);
    }
  }

  // Epilogue: C/D layout — lane col = lane&15, VGPR r holds M = r + (lane>>4)*8.
#pragma unroll
  for (int mi = 0; mi < 4; mi++) {
#pragma unroll
    for (int ni = 0; ni < 2; ni++) {
      int mbase = m0 + wm * 64 + mi * 16 + hh * 8;
      int n = n0 + wn * 32 + ni * 16 + ml;
#pragma unroll
      for (int r = 0; r < 8; r++) {
        float v = acc[mi][ni][r];
        size_t idx = ((size_t)e * M + (size_t)(mbase + r)) * Nn + n;
        if (ACT == 1) {
          v = (v > 0.f) ? v * v : 0.f;    // relu^2
          ((__bf16*)Out)[idx] = (__bf16)v;
        } else {
          ((float*)Out)[idx] = v;
        }
      }
    }
  }
}

// ---------------------------------------------------------------------------
__global__ __launch_bounds__(256) void combine_kernel(const float* __restrict__ eo,
                                                      const int* __restrict__ top_idx,
                                                      const int* __restrict__ rank,
                                                      const float* __restrict__ probs_m,
                                                      float* __restrict__ out) {
  int n = blockIdx.x;
  int e0 = top_idx[n * 2], e1 = top_idx[n * 2 + 1];
  int r0 = rank[n * 2];     if (r0 > CAP - 1) r0 = CAP - 1;
  int r1 = rank[n * 2 + 1]; if (r1 > CAP - 1) r1 = CAP - 1;
  float p0 = probs_m[n * 2], p1 = probs_m[n * 2 + 1];
  const float* a = eo + ((size_t)e0 * CAP + r0) * CDIM;
  const float* b = eo + ((size_t)e1 * CAP + r1) * CDIM;
  float* o = out + (size_t)n * CDIM;
  for (int c = threadIdx.x; c < CDIM; c += 256) o[c] = p0 * a[c] + p1 * b[c];
}

// ---------------------------------------------------------------------------
extern "C" void kernel_launch(void* const* d_in, const int* in_sizes, int n_in,
                              void* d_out, int out_size, void* d_ws, size_t ws_size,
                              hipStream_t stream) {
  (void)in_sizes; (void)n_in; (void)out_size; (void)ws_size;
  const float* x     = (const float*)d_in[0];   // [8,2048,1024]
  const float* wg    = (const float*)d_in[1];   // [8,1024]
  const float* cfc   = (const float*)d_in[2];   // [8,1024,4096]
  const float* cproj = (const float*)d_in[3];   // [8,4096,1024]
  float* out = (float*)d_out;

  char* ws = (char*)d_ws;
  size_t off = 0;
  auto alloc = [&](size_t bytes) -> char* {
    char* p = ws + off;
    off = (off + bytes + 255) & ~(size_t)255;
    return p;
  };
  int*    top_idx = (int*)alloc((size_t)N_TOK * 2 * sizeof(int));
  int*    rank    = (int*)alloc((size_t)N_TOK * 2 * sizeof(int));
  float*  probs   = (float*)alloc((size_t)N_TOK * 2 * sizeof(float));
  float*  probs_m = (float*)alloc((size_t)N_TOK * 2 * sizeof(float));
  __bf16* buf     = (__bf16*)alloc((size_t)NEXP * CAP * CDIM * 2);   // 84 MB
  __bf16* cfct    = (__bf16*)alloc((size_t)NEXP * CDIM * HDIM * 2);  // 67 MB, [E][4C][C]
  __bf16* cprojt  = (__bf16*)alloc((size_t)NEXP * HDIM * CDIM * 2);  // 67 MB, [E][C][4C]
  __bf16* hbuf    = (__bf16*)alloc((size_t)NEXP * CAP * HDIM * 2);   // 336 MB
  float*  eo      = (float*)alloc((size_t)NEXP * CAP * CDIM * 4);    // 168 MB

  router_kernel<<<N_TOK / 8, 256, 0, stream>>>(x, wg, top_idx, probs);
  rank_kernel<<<1, 1024, 0, stream>>>(top_idx, probs, rank, probs_m);
  long long n16 = (long long)NEXP * CAP * CDIM * 2 / 16;
  zero_kernel<<<(unsigned)((n16 + 255) / 256), 256, 0, stream>>>((uint4*)buf, n16);
  dispatch_kernel<<<N_TOK * 2, 256, 0, stream>>>(x, top_idx, rank, buf);
  transpose_cast_kernel<<<dim3(HDIM / 32, CDIM / 32, NEXP), 256, 0, stream>>>(cfc, cfct, CDIM, HDIM);
  transpose_cast_kernel<<<dim3(CDIM / 32, HDIM / 32, NEXP), 256, 0, stream>>>(cproj, cprojt, HDIM, CDIM);
  gemm_bf16_kernel<1><<<dim3(CAP / 128, HDIM / 128, NEXP), 256, 0, stream>>>(buf, cfct, (void*)hbuf, CAP, HDIM, CDIM);
  gemm_bf16_kernel<0><<<dim3(CAP / 128, CDIM / 128, NEXP), 256, 0, stream>>>(hbuf, cprojt, (void*)eo, CAP, CDIM, HDIM);
  combine_kernel<<<N_TOK, 256, 0, stream>>>(eo, top_idx, rank, probs_m, out);
}
